// Seq2Seq_266287973124
// MI455X (gfx1250) — compile-verified, matrix operand order checked by reading, not run on hardware
//
#include <hip/hip_runtime.h>
#include <hip/hip_bf16.h>

// ---------------------------------------------------------------------------
// Seq2Seq LSTM encoder/decoder with dot attention — CDNA5 (gfx1250) WMMA path
// B=32, S=256, T=64, D=H=512, V=32000, L=2
// Round 2: f16 activation buffers (no per-GEMM conversion/barrier on the
// serial path), 2 N-tiles per wave, async global->LDS staging (ASYNCcnt)
// for the hoisted big GEMMs where the A strip is shared by all waves.
// ---------------------------------------------------------------------------

#define B_  32
#define S_  256
#define T_  64
#define H_  512
#define D_  512
#define V_  32000
#define G4H (4 * H_)   // 2048

typedef _Float16 v16h __attribute__((ext_vector_type(16)));
typedef _Float16 v8h  __attribute__((ext_vector_type(8)));
typedef float    v8f  __attribute__((ext_vector_type(8)));

// ---------------------------------------------------------------------------
// Small/serial WMMA GEMM:  C[M x N](f32) = A[M x K](f16) @ Wfragᵀ (+bias)(+=C)
//   A: f16 row-major (leading dim lda) -> A fragments are two contiguous
//      16B loads per lane (ISA 16-bit A layout), no LDS, no barriers.
//   Wfrag: pre-packed f16 B-fragments, fragment (nt,kt) at (nt*Ktiles+kt)*512.
//   Each wave owns TWO adjacent N tiles: one A fragment feeds two WMMAs.
//   grid = (Nt/8, M/16), block = 128 (4 waves).
// ---------------------------------------------------------------------------
__global__ __launch_bounds__(128) void gemm_small(
    const _Float16* __restrict__ A, int lda,
    const _Float16* __restrict__ Wf, int Ktiles,
    const float* __restrict__ bias,
    float* __restrict__ C, long long ldc, int Nt, int accumulate)
{
    const int wave = threadIdx.x >> 5;
    const int lane = threadIdx.x & 31;
    const int nt0 = (blockIdx.x * 4 + wave) * 2;
    if (nt0 >= Nt) return;                    // wave-uniform: EXEC stays all-1s

    const int mrow0 = blockIdx.y * 16;
    const int rl = lane & 15;
    const int hi = lane >> 4;

    const _Float16* Arow = A + (size_t)(mrow0 + rl) * lda + hi * 8;
    const v16h* w0 = (const v16h*)(Wf + (size_t)nt0 * Ktiles * 512);
    const v16h* w1 = (const v16h*)(Wf + (size_t)(nt0 + 1) * Ktiles * 512);

    v8f acc0 = {}, acc1 = {};
    for (int kt = 0; kt < Ktiles; ++kt) {
        const v8h* pa = (const v8h*)(Arow + kt * 32);
        v8h a0 = pa[0];                       // K = kt*32 + hi*8 .. +7
        v8h a1 = pa[2];                       // K = kt*32 + 16 + hi*8 .. +7
        v16h a;
#pragma unroll
        for (int e = 0; e < 8; ++e) { a[e] = a0[e]; a[8 + e] = a1[e]; }

        v16h b0 = w0[(size_t)kt * 32 + lane];
        v16h b1 = w1[(size_t)kt * 32 + lane];
        if (kt + 1 < Ktiles) {
            __builtin_prefetch(&w0[(size_t)(kt + 1) * 32 + lane], 0, 3);
            __builtin_prefetch(&w1[(size_t)(kt + 1) * 32 + lane], 0, 3);
        }
        acc0 = __builtin_amdgcn_wmma_f32_16x16x32_f16(false, a, false, b0,
                                                      (short)0, acc0, false, false);
        acc1 = __builtin_amdgcn_wmma_f32_16x16x32_f16(false, a, false, b1,
                                                      (short)0, acc1, false, false);
    }

#pragma unroll
    for (int r = 0; r < 8; ++r) {
        int row = mrow0 + r + hi * 8;
        size_t i0 = (size_t)row * (size_t)ldc + nt0 * 16 + rl;
        size_t i1 = i0 + 16;
        float v0 = acc0[r], v1 = acc1[r];
        if (bias)       { v0 += bias[nt0 * 16 + rl]; v1 += bias[nt0 * 16 + 16 + rl]; }
        if (accumulate) { v0 += C[i0]; v1 += C[i1]; }
        C[i0] = v0; C[i1] = v1;
    }
}

// ---------------------------------------------------------------------------
// Big/hoisted WMMA GEMM (M large, K == 512, lda == K): the 16xK A strip is
// shared by all 4 waves -> stage it into LDS with the CDNA5 async copy
// engine (GLOBAL_LOAD_ASYNC_TO_LDS_B128, tracked by ASYNCcnt), then read
// A fragments with ds_load_b128 pairs. B fragments stream global->VGPR.
// ---------------------------------------------------------------------------
__global__ __launch_bounds__(128) void gemm_big(
    const _Float16* __restrict__ A,           // row-major, lda == Ktiles*32
    const _Float16* __restrict__ Wf, int Ktiles,
    const float* __restrict__ bias,
    float* __restrict__ C, long long ldc, int Nt)
{
    __shared__ _Float16 As[16 * 512];
    const int K = Ktiles * 32;                // 512 here
    const int mrow0 = blockIdx.y * 16;
    const _Float16* src = A + (size_t)mrow0 * K;   // strip is contiguous

    // async-stage 16*K halves (16KB) into LDS: 16B per async op per lane
    const unsigned lbase = (unsigned)(uintptr_t)&As[0];
    const int nchunks = (16 * K) / 8;         // 16-byte chunks
    for (int ck = threadIdx.x; ck < nchunks; ck += blockDim.x) {
        unsigned laddr = lbase + (unsigned)ck * 16u;
        unsigned long long gaddr = (unsigned long long)(uintptr_t)(src + (size_t)ck * 8);
        asm volatile("global_load_async_to_lds_b128 %0, %1, off"
                     :: "v"(laddr), "v"(gaddr) : "memory");
    }
    asm volatile("s_wait_asynccnt 0x0" ::: "memory");
    __syncthreads();

    const int wave = threadIdx.x >> 5;
    const int lane = threadIdx.x & 31;
    const int nt0 = (blockIdx.x * 4 + wave) * 2;
    if (nt0 >= Nt) return;
    const int rl = lane & 15;
    const int hi = lane >> 4;

    const v16h* w0 = (const v16h*)(Wf + (size_t)nt0 * Ktiles * 512);
    const v16h* w1 = (const v16h*)(Wf + (size_t)(nt0 + 1) * Ktiles * 512);

    v8f acc0 = {}, acc1 = {};
    for (int kt = 0; kt < Ktiles; ++kt) {
        const v8h* p0 = (const v8h*)&As[rl * K + kt * 32 + hi * 8];
        const v8h* p1 = (const v8h*)&As[rl * K + kt * 32 + 16 + hi * 8];
        v8h a0 = *p0, a1 = *p1;
        v16h a;
#pragma unroll
        for (int e = 0; e < 8; ++e) { a[e] = a0[e]; a[8 + e] = a1[e]; }

        v16h b0 = w0[(size_t)kt * 32 + lane];
        v16h b1 = w1[(size_t)kt * 32 + lane];
        acc0 = __builtin_amdgcn_wmma_f32_16x16x32_f16(false, a, false, b0,
                                                      (short)0, acc0, false, false);
        acc1 = __builtin_amdgcn_wmma_f32_16x16x32_f16(false, a, false, b1,
                                                      (short)0, acc1, false, false);
    }

#pragma unroll
    for (int r = 0; r < 8; ++r) {
        int row = mrow0 + r + hi * 8;
        size_t i0 = (size_t)row * (size_t)ldc + nt0 * 16 + rl;
        float v0 = acc0[r], v1 = acc1[r];
        if (bias) { v0 += bias[nt0 * 16 + rl]; v1 += bias[nt0 * 16 + 16 + rl]; }
        C[i0] = v0; C[i0 + 16] = v1;
    }
}

// ---------------------------------------------------------------------------
// Re-pack a row-major f32 weight W[N x K] into WMMA f16 B-fragments.
// ---------------------------------------------------------------------------
__global__ void pack_weights(const float* __restrict__ W, _Float16* __restrict__ out,
                             int K, int Ktiles, long long total)
{
    long long i = (long long)blockIdx.x * blockDim.x + threadIdx.x;
    if (i >= total) return;
    int pos = (int)(i & 511);
    long long f = i >> 9;
    int kt = (int)(f % Ktiles);
    int nt = (int)(f / Ktiles);
    int lane = pos >> 4, e = pos & 15;
    int koff = (e < 8 ? e : e + 8) + ((lane < 16) ? 0 : 8);
    int n = nt * 16 + (lane & 15);
    int k = kt * 32 + koff;
    out[i] = (_Float16)W[(size_t)n * K + k];
}

// ---------------------------------------------------------------------------
// Pointwise LSTM gates; maintains f32 cell state, emits f16 hidden copies
// (GEMM A operands) plus optional f32 hidden (attention/keys).
// ---------------------------------------------------------------------------
__device__ __forceinline__ float sigm_(float x) { return 1.0f / (1.0f + __expf(-x)); }

__global__ void lstm_pointwise(const float* __restrict__ G,
                               float* __restrict__ c,
                               _Float16* __restrict__ h16,
                               float* __restrict__ h32,     // nullable
                               _Float16* __restrict__ ys16) // nullable
{
    int idx = blockIdx.x * blockDim.x + threadIdx.x;
    if (idx >= B_ * H_) return;
    int b = idx >> 9, u = idx & (H_ - 1);
    const float* g = G + (size_t)b * G4H;
    float gi = g[u], gf = g[H_ + u], gg = g[2 * H_ + u], go = g[3 * H_ + u];
    float cn = sigm_(gf) * c[idx] + sigm_(gi) * tanhf(gg);
    float hn = sigm_(go) * tanhf(cn);
    c[idx] = cn;
    h16[idx] = (_Float16)hn;
    if (h32)  h32[idx] = hn;
    if (ys16) ys16[idx] = (_Float16)hn;
}

// ---------------------------------------------------------------------------
// Dot attention, one block per batch row. keys time-major (s*B+b, H) f32.
// Context written as f16 into xin_h[:, D:].
// ---------------------------------------------------------------------------
__global__ __launch_bounds__(256) void attention_kernel(
    const float* __restrict__ keys, const float* __restrict__ h1,
    const unsigned char* __restrict__ mask, _Float16* __restrict__ xin_h)
{
    __shared__ float wgt[S_];
    __shared__ float red[S_];
    int b = blockIdx.x;
    int s = threadIdx.x;

    const float* kr = keys + ((size_t)s * B_ + b) * H_;
    const float* hb = h1 + (size_t)b * H_;
    float sc = 0.f;
    for (int i = 0; i < H_; i += 4) {
        sc += kr[i] * hb[i] + kr[i + 1] * hb[i + 1]
            + kr[i + 2] * hb[i + 2] + kr[i + 3] * hb[i + 3];
    }
    if (!mask[b * S_ + s]) sc = -1e9f;

    red[s] = sc; __syncthreads();
    for (int off = 128; off; off >>= 1) {
        if (s < off) red[s] = fmaxf(red[s], red[s + off]);
        __syncthreads();
    }
    float m = red[0]; __syncthreads();
    float e = __expf(sc - m);
    red[s] = e; __syncthreads();
    for (int off = 128; off; off >>= 1) {
        if (s < off) red[s] += red[s + off];
        __syncthreads();
    }
    float inv = 1.0f / red[0];
    wgt[s] = e * inv;
    __syncthreads();

    for (int hh = s; hh < H_; hh += 256) {
        float acc = 0.f;
        for (int ss = 0; ss < S_; ++ss)
            acc += wgt[ss] * keys[((size_t)ss * B_ + b) * H_ + hh];
        xin_h[(size_t)b * (D_ + H_) + D_ + hh] = (_Float16)acc;
    }
}

// ---------------------------------------------------------------------------
// Embeddings (emit f16 GEMM operands directly), bias sums, zero fills
// ---------------------------------------------------------------------------
__global__ void embed_enc(const float* __restrict__ emb, const int* __restrict__ src,
                          _Float16* __restrict__ xs_h)
{
    long long i = (long long)blockIdx.x * blockDim.x + threadIdx.x;
    if (i >= (long long)S_ * B_ * D_) return;
    int d = (int)(i & (D_ - 1));
    int r = (int)(i >> 9);         // row = s*B + b
    int s = r >> 5;
    int b = r & (B_ - 1);
    int tok = src[b * S_ + s];
    xs_h[i] = (_Float16)emb[(size_t)tok * D_ + d];
}

__global__ void embed_dec(const float* __restrict__ emb, const int* __restrict__ dec,
                          int t, _Float16* __restrict__ xin_h)
{
    int i = blockIdx.x * blockDim.x + threadIdx.x;
    if (i >= B_ * D_) return;
    int b = i >> 9, d = i & (D_ - 1);
    int tok = dec[b * T_ + t];
    xin_h[(size_t)b * (D_ + H_) + d] = (_Float16)emb[(size_t)tok * D_ + d];
}

__global__ void bias_sums(const float* a0, const float* b0, const float* a1, const float* b1,
                          const float* a2, const float* b2, const float* a3, const float* b3,
                          float* out)
{
    int i = blockIdx.x * blockDim.x + threadIdx.x;
    if (i >= 4 * G4H) return;
    int w = i >> 11, j = i & (G4H - 1);
    const float* A[4] = {a0, a1, a2, a3};
    const float* Bb[4] = {b0, b1, b2, b3};
    out[i] = A[w][j] + Bb[w][j];
}

__global__ void zero_f32(float* p, long long n)
{
    long long i = (long long)blockIdx.x * blockDim.x + threadIdx.x;
    if (i < n) p[i] = 0.0f;
}

__global__ void zero_f16(_Float16* p, long long n)
{
    long long i = (long long)blockIdx.x * blockDim.x + threadIdx.x;
    if (i < n) p[i] = (_Float16)0.0f;
}

// ---------------------------------------------------------------------------
// Host orchestration
// ---------------------------------------------------------------------------
static inline int cdiv_(long long a, int b) { return (int)((a + b - 1) / b); }

extern "C" void kernel_launch(void* const* d_in, const int* in_sizes, int n_in,
                              void* d_out, int out_size, void* d_ws, size_t ws_size,
                              hipStream_t stream)
{
    (void)in_sizes; (void)n_in; (void)out_size; (void)ws_size;

    const float* enc_emb  = (const float*)d_in[0];
    const float* enc_Wih0 = (const float*)d_in[1];
    const float* enc_Whh0 = (const float*)d_in[2];
    const float* enc_bih0 = (const float*)d_in[3];
    const float* enc_bhh0 = (const float*)d_in[4];
    const float* enc_Wih1 = (const float*)d_in[5];
    const float* enc_Whh1 = (const float*)d_in[6];
    const float* enc_bih1 = (const float*)d_in[7];
    const float* enc_bhh1 = (const float*)d_in[8];
    const float* dec_emb  = (const float*)d_in[9];
    const float* dec_Wih0 = (const float*)d_in[10];
    const float* dec_Whh0 = (const float*)d_in[11];
    const float* dec_bih0 = (const float*)d_in[12];
    const float* dec_bhh0 = (const float*)d_in[13];
    const float* dec_Wih1 = (const float*)d_in[14];
    const float* dec_Whh1 = (const float*)d_in[15];
    const float* dec_bih1 = (const float*)d_in[16];
    const float* dec_bhh1 = (const float*)d_in[17];
    const float* out_W    = (const float*)d_in[18];
    const float* out_b    = (const float*)d_in[19];
    const int*   src      = (const int*)d_in[20];
    const unsigned char* src_mask = (const unsigned char*)d_in[21];
    const int*   dec_in_p = (const int*)d_in[22];
    float* out = (float*)d_out;

    // ---- workspace carve-up ----
    char* wpc = (char*)d_ws;
    auto alloc = [&](size_t bytes) -> void* {
        void* p = (void*)wpc;
        wpc += (bytes + 255) & ~(size_t)255;
        return p;
    };
    _Float16* wf_eWih0 = (_Float16*)alloc((size_t)G4H * D_ * 2);
    _Float16* wf_eWhh0 = (_Float16*)alloc((size_t)G4H * H_ * 2);
    _Float16* wf_eWih1 = (_Float16*)alloc((size_t)G4H * H_ * 2);
    _Float16* wf_eWhh1 = (_Float16*)alloc((size_t)G4H * H_ * 2);
    _Float16* wf_dWih0 = (_Float16*)alloc((size_t)G4H * (D_ + H_) * 2);
    _Float16* wf_dWhh0 = (_Float16*)alloc((size_t)G4H * H_ * 2);
    _Float16* wf_dWih1 = (_Float16*)alloc((size_t)G4H * H_ * 2);
    _Float16* wf_dWhh1 = (_Float16*)alloc((size_t)G4H * H_ * 2);
    _Float16* wf_outW  = (_Float16*)alloc((size_t)V_ * H_ * 2);
    float* bsum   = (float*)alloc(4 * G4H * sizeof(float));
    _Float16* xs_h  = (_Float16*)alloc((size_t)S_ * B_ * D_ * 2);
    _Float16* ys0_h = (_Float16*)alloc((size_t)S_ * B_ * H_ * 2);
    float* P      = (float*)alloc((size_t)S_ * B_ * G4H * sizeof(float));
    float* keys   = (float*)alloc((size_t)S_ * B_ * H_ * sizeof(float));
    float* c0     = (float*)alloc((size_t)B_ * H_ * sizeof(float));
    float* c1     = (float*)alloc((size_t)B_ * H_ * sizeof(float));
    float* h1f    = (float*)alloc((size_t)B_ * H_ * sizeof(float));
    _Float16* h0_h  = (_Float16*)alloc((size_t)B_ * H_ * 2);
    _Float16* h1_h  = (_Float16*)alloc((size_t)B_ * H_ * 2);
    _Float16* xin_h = (_Float16*)alloc((size_t)B_ * (D_ + H_) * 2);
    float* G      = (float*)alloc((size_t)B_ * G4H * sizeof(float));

    // ---- init state every call (harness does not re-poison) ----
    {
        long long n = (long long)B_ * H_;
        zero_f32<<<cdiv_(n, 256), 256, 0, stream>>>(c0, n);
        zero_f32<<<cdiv_(n, 256), 256, 0, stream>>>(c1, n);
        zero_f16<<<cdiv_(n, 256), 256, 0, stream>>>(h0_h, n);
        zero_f16<<<cdiv_(n, 256), 256, 0, stream>>>(h1_h, n);
        long long nx = (long long)B_ * (D_ + H_);
        zero_f16<<<cdiv_(nx, 256), 256, 0, stream>>>(xin_h, nx);
    }

    bias_sums<<<cdiv_(4 * G4H, 256), 256, 0, stream>>>(
        enc_bih0, enc_bhh0, enc_bih1, enc_bhh1,
        dec_bih0, dec_bhh0, dec_bih1, dec_bhh1, bsum);
    float* bs_e0 = bsum;            float* bs_e1 = bsum + G4H;
    float* bs_d0 = bsum + 2 * G4H;  float* bs_d1 = bsum + 3 * G4H;

    auto pack = [&](const float* W, _Float16* dst, int K, int N) {
        int Kt = K / 32;
        long long total = (long long)N * K;
        pack_weights<<<cdiv_(total, 256), 256, 0, stream>>>(W, dst, K, Kt, total);
    };
    pack(enc_Wih0, wf_eWih0, D_, G4H);
    pack(enc_Whh0, wf_eWhh0, H_, G4H);
    pack(enc_Wih1, wf_eWih1, H_, G4H);
    pack(enc_Whh1, wf_eWhh1, H_, G4H);
    pack(dec_Wih0, wf_dWih0, D_ + H_, G4H);
    pack(dec_Whh0, wf_dWhh0, H_, G4H);
    pack(dec_Wih1, wf_dWih1, H_, G4H);
    pack(dec_Whh1, wf_dWhh1, H_, G4H);
    pack(out_W,    wf_outW,  H_, V_);

    {
        long long n = (long long)S_ * B_ * D_;
        embed_enc<<<cdiv_(n, 256), 256, 0, stream>>>(enc_emb, src, xs_h);
    }

    const int NtG = G4H / 16;      // 128
    const int NtV = V_ / 16;       // 2000
    const dim3 blk(128);

    // ---- encoder layer 0: hoisted input GEMM (async-LDS), then scan ----
    gemm_big<<<dim3(NtG / 8, (S_ * B_) / 16), blk, 0, stream>>>(
        xs_h, wf_eWih0, D_ / 32, bs_e0, P, G4H, NtG);
    for (int s = 0; s < S_; ++s) {
        float* Ps = P + (size_t)s * B_ * G4H;
        gemm_small<<<dim3(NtG / 8, B_ / 16), blk, 0, stream>>>(
            h0_h, H_, wf_eWhh0, H_ / 32, nullptr, Ps, G4H, NtG, 1);
        lstm_pointwise<<<cdiv_(B_ * H_, 256), 256, 0, stream>>>(
            Ps, c0, h0_h, nullptr, ys0_h + (size_t)s * B_ * H_);
    }

    // ---- encoder layer 1 ----
    gemm_big<<<dim3(NtG / 8, (S_ * B_) / 16), blk, 0, stream>>>(
        ys0_h, wf_eWih1, H_ / 32, bs_e1, P, G4H, NtG);
    for (int s = 0; s < S_; ++s) {
        float* Ps = P + (size_t)s * B_ * G4H;
        gemm_small<<<dim3(NtG / 8, B_ / 16), blk, 0, stream>>>(
            h1_h, H_, wf_eWhh1, H_ / 32, nullptr, Ps, G4H, NtG, 1);
        lstm_pointwise<<<cdiv_(B_ * H_, 256), 256, 0, stream>>>(
            Ps, c1, h1_h, keys + (size_t)s * B_ * H_, nullptr);
    }

    // ---- decoder ----
    auto dec_cell = [&]() {
        gemm_small<<<dim3(NtG / 8, B_ / 16), blk, 0, stream>>>(
            xin_h, D_ + H_, wf_dWih0, (D_ + H_) / 32, bs_d0, G, G4H, NtG, 0);
        gemm_small<<<dim3(NtG / 8, B_ / 16), blk, 0, stream>>>(
            h0_h, H_, wf_dWhh0, H_ / 32, nullptr, G, G4H, NtG, 1);
        lstm_pointwise<<<cdiv_(B_ * H_, 256), 256, 0, stream>>>(
            G, c0, h0_h, nullptr, nullptr);
        gemm_small<<<dim3(NtG / 8, B_ / 16), blk, 0, stream>>>(
            h0_h, H_, wf_dWih1, H_ / 32, bs_d1, G, G4H, NtG, 0);
        gemm_small<<<dim3(NtG / 8, B_ / 16), blk, 0, stream>>>(
            h1_h, H_, wf_dWhh1, H_ / 32, nullptr, G, G4H, NtG, 1);
        lstm_pointwise<<<cdiv_(B_ * H_, 256), 256, 0, stream>>>(
            G, c1, h1_h, h1f, nullptr);
    };

    for (int t = 0; t < T_; ++t) {
        embed_dec<<<cdiv_(B_ * D_, 256), 256, 0, stream>>>(dec_emb, dec_in_p, t, xin_h);
        dec_cell();                                        // first decode step
        attention_kernel<<<B_, 256, 0, stream>>>(keys, h1f, src_mask, xin_h);
        dec_cell();                                        // second decode step
        gemm_small<<<dim3(NtV / 8, B_ / 16), blk, 0, stream>>>(
            h1_h, H_, wf_outW, H_ / 32, out_b,
            out + (size_t)t * V_, (long long)T_ * V_, NtV, 0);
    }
}
